// SiameseGAT_v3_88751204205243
// MI455X (gfx1250) — matrix operand order repaired and checked by reading, not compiled, not run on hardware
//
#include <hip/hip_runtime.h>
#include <hip/hip_bf16.h>

typedef __attribute__((ext_vector_type(16))) _Float16 v16h;
typedef __attribute__((ext_vector_type(8)))  _Float16 v8h;
typedef __attribute__((ext_vector_type(8)))  float    v8f;

#define NN   20000
#define NE   320000
#define NG   32
#define INC  5
#define HID  64
#define NH   4
#define DH   16
#define NL   8
#define ETOT (NE + NN)      // 340000 (self-loops appended)
#define MTILES (NN / 16)    // 1250 exact

// ---- ordered-uint encoding for float atomicMax (handles negatives) ----
__device__ __forceinline__ unsigned encOrd(float f) {
  int i = __float_as_int(f);
  return (i >= 0) ? ((unsigned)i ^ 0x80000000u) : ~(unsigned)i;
}
__device__ __forceinline__ float decOrd(unsigned u) {
  int i = (u & 0x80000000u) ? (int)(u ^ 0x80000000u) : (int)~u;
  return __int_as_float(i);
}

__global__ void k_zero_f(float* p, int n) {
  int i = blockIdx.x * blockDim.x + threadIdx.x;
  if (i < n) p[i] = 0.f;
}

// Pre-swizzle Wl/Wr into CDNA5 16-bit B-fragment layout.
// idx = ((((l*2+side)*4 + nt)*2 + kt)*32 + lane)*16 + j
// lane 0-15: n = nt*16+lane,   K [0..7]+[16..23] of chunk
// lane16-31: n = nt*16+lane-16, K [8..15]+[24..31] of chunk
__global__ void k_wfrag(const float* __restrict__ Wl, const float* __restrict__ Wr,
                        _Float16* __restrict__ wfrag) {
  int idx = blockIdx.x * blockDim.x + threadIdx.x;
  if (idx >= NL * 2 * 4 * 2 * 32 * 16) return;
  int j    = idx & 15;
  int lane = (idx >> 4) & 31;
  int kt   = (idx >> 9) & 1;
  int nt   = (idx >> 10) & 3;
  int side = (idx >> 12) & 1;
  int l    = idx >> 13;
  const float* W = (side == 0 ? Wl : Wr) + (size_t)l * HID * HID;
  int n = nt * 16 + (lane & 15);
  int k = kt * 32 + ((j & 8) ? 16 : 0) + (lane >> 4) * 8 + (j & 7);
  wfrag[idx] = (_Float16)W[k * HID + n];
}

__global__ void k_node_lin(const float* __restrict__ x, const float* __restrict__ Wn,
                           const float* __restrict__ bn, float* __restrict__ h) {
  int i = blockIdx.x * blockDim.x + threadIdx.x;
  if (i >= NN * HID) return;
  int node = i >> 6, c = i & 63;
  float s = bn[c];
#pragma unroll
  for (int k = 0; k < INC; ++k) s += x[node * INC + k] * Wn[k * HID + c];
  h[i] = s;
}

__global__ void k_f16cvt(const float* __restrict__ h, _Float16* __restrict__ hf) {
  int i = blockIdx.x * blockDim.x + threadIdx.x;
  if (i < NN * HID) hf[i] = (_Float16)h[i];
}

// One wave computes a 16x64 output tile of BOTH xl and xr (A reused).
// A layout (16-bit 16x32): lane 0-15 row=lane holds K [0..7]+[16..23];
// lane 16-31 row=lane-16 holds K [8..15]+[24..31]  (per k-chunk of 32).
__global__ void k_gemm(const _Float16* __restrict__ hf,
                       const _Float16* __restrict__ wfragL, const _Float16* __restrict__ wfragR,
                       const float* __restrict__ bl, const float* __restrict__ br,
                       float* __restrict__ xl, float* __restrict__ xr) {
  int wave = (blockIdx.x * blockDim.x + threadIdx.x) >> 5;
  int lane = threadIdx.x & 31;
  if (wave >= MTILES) return;     // wave-uniform: EXEC stays all-ones for WMMA
  int m0 = wave << 4;
  int row = m0 + (lane & 15);
  int kb = (lane >> 4) << 3;      // 0 or 8
  const _Float16* hrow = hf + (size_t)row * HID;
  v8h a0lo = *(const v8h*)(hrow + kb);
  v8h a0hi = *(const v8h*)(hrow + kb + 16);
  v8h a1lo = *(const v8h*)(hrow + 32 + kb);
  v8h a1hi = *(const v8h*)(hrow + 32 + kb + 16);
  v16h a0 = __builtin_shufflevector(a0lo, a0hi, 0,1,2,3,4,5,6,7,8,9,10,11,12,13,14,15);
  v16h a1 = __builtin_shufflevector(a1lo, a1hi, 0,1,2,3,4,5,6,7,8,9,10,11,12,13,14,15);
  int col = lane & 15;
  int rbase = m0 + ((lane >> 4) << 3);   // D: VGPR r -> row m0 + (lane>=16?8:0) + r
#pragma unroll
  for (int side = 0; side < 2; ++side) {
    const _Float16* wf = side ? wfragR : wfragL;
    const float* bias  = side ? br : bl;
    float* out         = side ? xr : xl;
#pragma unroll
    for (int nt = 0; nt < 4; ++nt) {
      v16h b0 = *(const v16h*)(wf + ((size_t)(nt * 2 + 0) * 32 + lane) * 16);
      v16h b1 = *(const v16h*)(wf + ((size_t)(nt * 2 + 1) * 32 + lane) * 16);
      v8f c = {};
      c = __builtin_amdgcn_wmma_f32_16x16x32_f16(false, a0, false, b0, (short)0, c, false, false);
      c = __builtin_amdgcn_wmma_f32_16x16x32_f16(false, a1, false, b1, (short)0, c, false, false);
      float bv = bias[nt * 16 + col];
#pragma unroll
      for (int r = 0; r < 8; ++r)
        out[(size_t)(rbase + r) * HID + nt * 16 + col] = c[r] + bv;
    }
  }
}

__global__ void k_init_layer(float* outacc, unsigned* smax, float* denom, float* bnsum) {
  int i = blockIdx.x * blockDim.x + threadIdx.x;
  if (i < NN * HID) outacc[i] = 0.f;
  if (i < NN * NH) { smax[i] = 0u; denom[i] = 0.f; }  // 0 < encOrd(-inf): lower bound
  if (i < 2 * HID) bnsum[i] = 0.f;
}

__global__ void k_edge_score(const int* __restrict__ ei, const float* __restrict__ xl,
                             const float* __restrict__ xr, const float* __restrict__ att,
                             float* __restrict__ score, unsigned* __restrict__ smax) {
  int e = blockIdx.x * blockDim.x + threadIdx.x;
  if (e >= ETOT) return;
  int src, dst;
  if (e < NE) { src = ei[e]; dst = ei[NE + e]; } else { src = dst = e - NE; }
  const float* pl = xl + (size_t)src * HID;
  const float* pr = xr + (size_t)dst * HID;
#pragma unroll
  for (int h = 0; h < NH; ++h) {
    float s = 0.f;
#pragma unroll
    for (int d = 0; d < DH; ++d) {
      float v = pl[h * DH + d] + pr[h * DH + d];
      v = v > 0.f ? v : 0.2f * v;              // leaky_relu(0.2)
      s += att[h * DH + d] * v;
    }
    score[(size_t)e * NH + h] = s;
    atomicMax(&smax[dst * NH + h], encOrd(s));
  }
}

__global__ void k_edge_exp(const int* __restrict__ ei, const unsigned* __restrict__ smax,
                           float* __restrict__ score, float* __restrict__ denom) {
  int e = blockIdx.x * blockDim.x + threadIdx.x;
  if (e >= ETOT) return;
  int dst = (e < NE) ? ei[NE + e] : e - NE;
#pragma unroll
  for (int h = 0; h < NH; ++h) {
    float m = decOrd(smax[dst * NH + h]);
    float ex = __expf(score[(size_t)e * NH + h] - m);
    score[(size_t)e * NH + h] = ex;
    atomicAdd(&denom[dst * NH + h], ex);
  }
}

__global__ void k_edge_scatter(const int* __restrict__ ei, const float* __restrict__ xl,
                               const float* __restrict__ score, const float* __restrict__ denom,
                               float* __restrict__ outacc) {
  int e = blockIdx.x * blockDim.x + threadIdx.x;
  if (e >= ETOT) return;
  int src, dst;
  if (e < NE) { src = ei[e]; dst = ei[NE + e]; } else { src = dst = e - NE; }
  const float* pl = xl + (size_t)src * HID;
#pragma unroll
  for (int h = 0; h < NH; ++h) {
    float alpha = score[(size_t)e * NH + h] / denom[dst * NH + h];
#pragma unroll
    for (int d = 0; d < DH; ++d)
      atomicAdd(&outacc[(size_t)dst * HID + h * DH + d], alpha * pl[h * DH + d]);
  }
}

// blockDim = 64 (one channel per thread), grid = 256 partial blocks
__global__ void k_bn_stats(const float* __restrict__ outacc, const float* __restrict__ cbias,
                           float* __restrict__ bnsum) {
  int ch = threadIdx.x;
  float s = 0.f, sq = 0.f;
  float cb = cbias[ch];
  for (int r = blockIdx.x; r < NN; r += gridDim.x) {
    float g = outacc[(size_t)r * HID + ch] + cb;
    s += g; sq += g * g;
  }
  atomicAdd(&bnsum[ch], s);
  atomicAdd(&bnsum[HID + ch], sq);
}

__global__ void k_bn_apply(const float* __restrict__ outacc, const float* __restrict__ cbias,
                           const float* __restrict__ bnsum, const float* __restrict__ gamma,
                           const float* __restrict__ beta, float* __restrict__ h) {
  int i = blockIdx.x * blockDim.x + threadIdx.x;
  if (i >= NN * HID) return;
  int ch = i & 63;
  float mu = bnsum[ch] * (1.f / NN);
  float var = bnsum[HID + ch] * (1.f / NN) - mu * mu;
  var = var > 0.f ? var : 0.f;
  float g = (outacc[i] + cbias[ch] - mu) * rsqrtf(var + 1e-5f) * gamma[ch] + beta[ch];
  float el = g > 0.f ? g : (__expf(g) - 1.f);   // elu
  h[i] += el;
}

__global__ void k_pool(const float* __restrict__ h, const int* __restrict__ batch,
                       float* __restrict__ pool, float* __restrict__ cnt) {
  int i = blockIdx.x * blockDim.x + threadIdx.x;
  if (i >= NN * HID) return;
  int node = i >> 6, c = i & 63;
  int b = batch[node];
  atomicAdd(&pool[(size_t)b * HID + c], h[i]);
  if (c == 0) atomicAdd(&cnt[b], 1.f);
}

// Tiny FC head: one block, staged through LDS (48 KB).
__global__ void k_head(const float* __restrict__ pool1, const float* __restrict__ cnt1,
                       const float* __restrict__ pool2, const float* __restrict__ cnt2,
                       const float* __restrict__ f1w, const float* __restrict__ f1b,
                       const float* __restrict__ f2w, const float* __restrict__ f2b,
                       const float* __restrict__ f3w, const float* __restrict__ f3b,
                       float* __restrict__ out) {
  __shared__ float sc[NG * 128];
  __shared__ float sh1[NG * 256];
  int t = threadIdx.x;
  for (int i = t; i < NG * 128; i += 256) {
    int g = i >> 7, c = i & 127;
    float v;
    if (c < 64) { float n = cnt1[g]; n = n > 1.f ? n : 1.f; v = pool1[g * 64 + c] / n; }
    else        { float n = cnt2[g]; n = n > 1.f ? n : 1.f; v = pool2[g * 64 + (c - 64)] / n; }
    sc[i] = v;
  }
  __syncthreads();
  for (int i = t; i < NG * 256; i += 256) {
    int g = i >> 8, j = i & 255;
    float s = f1b[j];
    for (int k = 0; k < 128; ++k) s += sc[g * 128 + k] * f1w[k * 256 + j];
    sh1[i] = s > 0.f ? s : 0.f;
  }
  __syncthreads();
  for (int i = t; i < NG * 128; i += 256) {
    int g = i >> 7, j = i & 127;
    float s = f2b[j];
    for (int k = 0; k < 256; ++k) s += sh1[g * 256 + k] * f2w[k * 128 + j];
    sc[i] = s > 0.f ? s : 0.f;
  }
  __syncthreads();
  if (t < NG) {
    float s = f3b[0];
    for (int k = 0; k < 128; ++k) s += sc[t * 128 + k] * f3w[k];
    out[t] = s;
  }
}

extern "C" void kernel_launch(void* const* d_in, const int* in_sizes, int n_in,
                              void* d_out, int out_size, void* d_ws, size_t ws_size,
                              hipStream_t stream) {
  (void)in_sizes; (void)n_in; (void)out_size; (void)ws_size;
  const float* x1    = (const float*)d_in[0];
  const float* x2    = (const float*)d_in[1];
  const int*   ei1   = (const int*)d_in[2];
  const int*   ei2   = (const int*)d_in[3];
  const int*   bat1  = (const int*)d_in[4];
  const int*   bat2  = (const int*)d_in[5];
  const float* Wn    = (const float*)d_in[6];
  const float* bnode = (const float*)d_in[7];
  const float* Wl    = (const float*)d_in[8];
  const float* bl    = (const float*)d_in[9];
  const float* Wr    = (const float*)d_in[10];
  const float* br    = (const float*)d_in[11];
  const float* att   = (const float*)d_in[12];
  const float* cbias = (const float*)d_in[13];
  const float* gamma = (const float*)d_in[14];
  const float* beta  = (const float*)d_in[15];
  const float* f1w   = (const float*)d_in[16];
  const float* f1b   = (const float*)d_in[17];
  const float* f2w   = (const float*)d_in[18];
  const float* f2b   = (const float*)d_in[19];
  const float* f3w   = (const float*)d_in[20];
  const float* f3b   = (const float*)d_in[21];

  char* ws = (char*)d_ws;
  size_t off = 0;
  auto carve = [&](size_t bytes) -> char* {
    char* p = ws + off;
    off += (bytes + 255) & ~(size_t)255;
    return p;
  };
  _Float16* wfrag  = (_Float16*)carve((size_t)NL * 2 * 4 * 2 * 32 * 16 * sizeof(_Float16));
  float*    h      = (float*)carve((size_t)NN * HID * 4);
  _Float16* hf     = (_Float16*)carve((size_t)NN * HID * 2);
  float*    xl     = (float*)carve((size_t)NN * HID * 4);
  float*    xr     = (float*)carve((size_t)NN * HID * 4);
  float*    outacc = (float*)carve((size_t)NN * HID * 4);
  float*    score  = (float*)carve((size_t)ETOT * NH * 4);
  unsigned* smax   = (unsigned*)carve((size_t)NN * NH * 4);
  float*    denom  = (float*)carve((size_t)NN * NH * 4);
  float*    bnsum  = (float*)carve(2 * HID * 4);
  float*    pool1  = (float*)carve((size_t)NG * HID * 4);
  float*    cnt1   = (float*)carve(NG * 4);
  float*    pool2  = (float*)carve((size_t)NG * HID * 4);
  float*    cnt2   = (float*)carve(NG * 4);

  const int TB = 256;
  int gNH = (NN * HID + TB - 1) / TB;                // 5000
  int gE  = (ETOT + TB - 1) / TB;                    // 1329
  int gW  = (NL * 2 * 4 * 2 * 32 * 16 + TB - 1) / TB;
  int gG  = (MTILES * 32 + TB - 1) / TB;             // 157 (8 waves/block)

  k_wfrag<<<gW, TB, 0, stream>>>(Wl, Wr, wfrag);
  k_zero_f<<<(NG * HID + TB - 1) / TB, TB, 0, stream>>>(pool1, NG * HID);
  k_zero_f<<<1, TB, 0, stream>>>(cnt1, NG);
  k_zero_f<<<(NG * HID + TB - 1) / TB, TB, 0, stream>>>(pool2, NG * HID);
  k_zero_f<<<1, TB, 0, stream>>>(cnt2, NG);

  for (int pass = 0; pass < 2; ++pass) {
    const float* x  = pass ? x2 : x1;
    const int*   ei = pass ? ei2 : ei1;
    const int*   bt = pass ? bat2 : bat1;
    float* pool = pass ? pool2 : pool1;
    float* cnt  = pass ? cnt2 : cnt1;

    k_node_lin<<<gNH, TB, 0, stream>>>(x, Wn, bnode, h);
    for (int l = 0; l < NL; ++l) {
      k_f16cvt<<<gNH, TB, 0, stream>>>(h, hf);
      k_gemm<<<gG, TB, 0, stream>>>(hf,
          wfrag + (size_t)(l * 2 + 0) * 4 * 2 * 32 * 16,
          wfrag + (size_t)(l * 2 + 1) * 4 * 2 * 32 * 16,
          bl + l * HID, br + l * HID, xl, xr);
      k_init_layer<<<gNH, TB, 0, stream>>>(outacc, smax, denom, bnsum);
      k_edge_score<<<gE, TB, 0, stream>>>(ei, xl, xr, att + l * NH * DH, score, smax);
      k_edge_exp<<<gE, TB, 0, stream>>>(ei, smax, score, denom);
      k_edge_scatter<<<gE, TB, 0, stream>>>(ei, xl, score, denom, outacc);
      k_bn_stats<<<256, 64, 0, stream>>>(outacc, cbias + l * HID, bnsum);
      k_bn_apply<<<gNH, TB, 0, stream>>>(outacc, cbias + l * HID, bnsum,
                                         gamma + l * HID, beta + l * HID, h);
    }
    k_pool<<<gNH, TB, 0, stream>>>(h, bt, pool, cnt);
  }
  k_head<<<1, 256, 0, stream>>>(pool1, cnt1, pool2, cnt2,
                                f1w, f1b, f2w, f2b, f3w, f3b, (float*)d_out);
}
